// GQA_65326452572679
// MI455X (gfx1250) — matrix operand order repaired and checked by reading, not compiled, not run on hardware
//
#include <hip/hip_runtime.h>
#include <hip/hip_bf16.h>
#include <math.h>

// ---------------------------------------------------------------------------
// GQA forward for MI455X (gfx1250, wave32, WMMA 16x16x32 f16 -> f32 acc)
// ---------------------------------------------------------------------------

typedef __attribute__((ext_vector_type(16))) _Float16 v16h;
typedef __attribute__((ext_vector_type(8)))  _Float16 v8h;
typedef __attribute__((ext_vector_type(8)))  float    v8f;

#define DMODEL 2048
#define TSEQ   2048
#define BATCH  2
#define NHEADS 32
#define NKV    8
#define HDIM   64
#define KVDIM  (NKV * HDIM)   // 512
#define ROWS   (BATCH * TSEQ) // 4096

static __device__ inline v8f zero8() {
  v8f z;
#pragma unroll
  for (int e = 0; e < 8; ++e) z[e] = 0.0f;
  return z;
}

// A-fragment (16x32 f16, MxK). Lane l: row m = l&15, ksel = l>>4.
// elements 0..15 = K = 16*ksel .. 16*ksel+15 (contiguous, matches B frags)
static __device__ inline v16h load_afrag(const _Float16* rowk, int ksel) {
  v8h lo = *(const v8h*)(rowk + 16 * ksel);
  v8h hi = *(const v8h*)(rowk + 16 * ksel + 8);
  v16h a;
#pragma unroll
  for (int e = 0; e < 8; ++e) { a[e] = lo[e]; a[e + 8] = hi[e]; }
  return a;
}

static __device__ inline v8f wmma_f16(v16h a, v16h b, v8f c) {
  return __builtin_amdgcn_wmma_f32_16x16x32_f16(
      /*neg_a=*/false, a, /*neg_b=*/false, b,
      /*c_mod=*/(short)0, c, /*reuse_a=*/false, /*reuse_b=*/false);
}

// ---------------------------------------------------------------------------
__global__ void cvt_f32_f16(const float* __restrict__ in,
                            _Float16* __restrict__ out, long long n) {
  long long i = (long long)blockIdx.x * blockDim.x + threadIdx.x;
  long long stride = (long long)gridDim.x * blockDim.x;
  for (; i < n; i += stride) out[i] = (_Float16)in[i];
}

// Vt[b][hkv][d][t] = V[b][t][hkv*64+d]
__global__ void vtrans_kernel(const _Float16* __restrict__ v,
                              _Float16* __restrict__ vt) {
  long long i = (long long)blockIdx.x * blockDim.x + threadIdx.x;
  if (i >= (long long)ROWS * KVDIM) return;
  int d   = (int)(i & 63);
  int hkv = (int)((i >> 6) & 7);
  int t   = (int)((i >> 9) & 2047);
  int b   = (int)(i >> 20);
  vt[(((long long)(b * NKV + hkv)) * HDIM + d) * TSEQ + t] = v[i];
}

// ---------------------------------------------------------------------------
// C[M,N] = A[M,K] * W[N,K]^T.  A,W f16 row-major (K contiguous).
// Block = 128 threads = 4 waves in 2x2 -> block tile 128(M) x 64(N).
// Wave tile = 64(M) x 32(N) = 4x2 WMMA tiles, register double-buffered over K.
// B-fragment (32x16, KxN): lane l holds column n = l&15, K = 16*(l>>4)+0..15,
// i.e. 16 contiguous halfs of W row (n0+n).                     (this code's k order, same as A frags)
template <typename OutT>
__global__ __launch_bounds__(128) void gemm_f16_wmma(
    const _Float16* __restrict__ A, const _Float16* __restrict__ W,
    OutT* __restrict__ C, int M, int N, int K) {
  const int lane = threadIdx.x & 31;
  const int wave = threadIdx.x >> 5;
  const int wm = wave >> 1;
  const int wn = wave & 1;
  const int n0 = blockIdx.x * 64 + wn * 32;
  const int m0 = blockIdx.y * 128 + wm * 64;
  const int nl = lane & 15;
  const int ksel = lane >> 4;
  const int mb = 8 * ksel;

  const _Float16* wrow0 = W + (size_t)(n0 + nl) * K;
  const _Float16* wrow1 = W + (size_t)(n0 + 16 + nl) * K;
  const _Float16* arow[4];
#pragma unroll
  for (int i = 0; i < 4; ++i) arow[i] = A + (size_t)(m0 + i * 16 + nl) * K;

  auto load_tiles = [&](int k0, v16h a[4], v16h b[2]) {
    b[0] = *(const v16h*)(wrow0 + k0 + 16 * ksel);
    b[1] = *(const v16h*)(wrow1 + k0 + 16 * ksel);
#pragma unroll
    for (int i = 0; i < 4; ++i) a[i] = load_afrag(arow[i] + k0, ksel);
  };

  v8f acc[4][2];
#pragma unroll
  for (int i = 0; i < 4; ++i) { acc[i][0] = zero8(); acc[i][1] = zero8(); }

  v16h a_c[4], b_c[2];
  load_tiles(0, a_c, b_c);

  for (int k0 = 0; k0 < K; k0 += 32) {
    // prefetch one step further ahead (gfx1250 global_prefetch_b8)
    __builtin_prefetch(wrow0 + k0 + 64, 0, 0);
    __builtin_prefetch(arow[lane >> 3 & 3] + k0 + 64, 0, 0);
    // double-buffer: issue next step's loads before consuming current frags
    const int kn = (k0 + 32 < K) ? (k0 + 32) : k0;
    v16h a_n[4], b_n[2];
    load_tiles(kn, a_n, b_n);
#pragma unroll
    for (int i = 0; i < 4; ++i)
#pragma unroll
      for (int j = 0; j < 2; ++j)
        acc[i][j] = wmma_f16(a_c[i], b_c[j], acc[i][j]);
#pragma unroll
    for (int i = 0; i < 4; ++i) a_c[i] = a_n[i];
    b_c[0] = b_n[0];
    b_c[1] = b_n[1];
  }
  // C/D layout: lane l -> col +(l&15); element r -> row +8*(l>>4)+r
#pragma unroll
  for (int i = 0; i < 4; ++i)
#pragma unroll
    for (int j = 0; j < 2; ++j)
#pragma unroll
      for (int r = 0; r < 8; ++r) {
        size_t row = (size_t)(m0 + i * 16 + mb + r);
        C[row * (size_t)N + (n0 + j * 16 + nl)] = (OutT)acc[i][j][r];
      }
}

// ---------------------------------------------------------------------------
// Flash attention. One wave owns a 16-query tile of one head; key blocks of 64.
// Q: [B*T, 2048] (head h at col h*64), K: [B*T, 512], Vt: [B][hkv][64][T],
// ctx: [B*T, 2048] f16.
__global__ __launch_bounds__(128) void gqa_attn(
    const _Float16* __restrict__ Q, const _Float16* __restrict__ Km,
    const _Float16* __restrict__ Vt, _Float16* __restrict__ ctx) {
  __shared__ __align__(64) float    Sf[4][16][64];
  __shared__ __align__(64) _Float16 Pf[4][16][64];
  __shared__ __align__(64) float    Alpha[4][16];
  __shared__ __align__(64) float    Linv[4][16];

  const int lane = threadIdx.x & 31;
  const int wave = threadIdx.x >> 5;
  const int tile = blockIdx.x * 4 + wave;     // 8192 tiles total
  const int qt  = tile & 127;                 // T/16 = 128
  const int h   = (tile >> 7) & 31;
  const int b   = tile >> 12;
  const int hkv = h >> 2;
  const int nl = lane & 15;
  const int ksel = lane >> 4;
  const int mb = 8 * ksel;

  // Q fragments for this 16-row tile: rows qt*16+nl, dims 0..31 and 32..63
  const _Float16* qrow =
      Q + ((size_t)(b * TSEQ + qt * 16 + nl)) * DMODEL + h * HDIM;
  v16h qf[2];
#pragma unroll
  for (int c = 0; c < 2; ++c) qf[c] = load_afrag(qrow + 32 * c, ksel);

  const _Float16* kbase = Km + ((size_t)b * TSEQ) * KVDIM + hkv * HDIM;
  const _Float16* vtb = Vt + ((size_t)(b * NKV + hkv)) * HDIM * TSEQ;

  v8f o[4];
#pragma unroll
  for (int f = 0; f < 4; ++f) o[f] = zero8();
  float m_run = -3.0e38f, l_run = 0.0f;  // meaningful in lanes 0..15

  for (int j0 = 0; j0 < TSEQ; j0 += 64) {
    // prefetch next key/value block while computing this one
    if (j0 + 64 < TSEQ) {
      __builtin_prefetch(kbase + (size_t)(j0 + 64 + lane) * KVDIM, 0, 0);
      __builtin_prefetch(kbase + (size_t)(j0 + 96 + lane) * KVDIM, 0, 0);
      __builtin_prefetch(vtb + (size_t)(lane * 2) * TSEQ + j0 + 64, 0, 0);
    }
    // ---- batch-load all K fragments (8), then S = Q * K_blk^T (16 x 64)
    v16h kf[8];
#pragma unroll
    for (int nh = 0; nh < 4; ++nh) {
      const _Float16* krow = kbase + (size_t)(j0 + nh * 16 + nl) * KVDIM;
      kf[nh * 2 + 0] = *(const v16h*)(krow + 16 * ksel);
      kf[nh * 2 + 1] = *(const v16h*)(krow + 32 + 16 * ksel);
    }
    v8f s[4];
#pragma unroll
    for (int nh = 0; nh < 4; ++nh) s[nh] = zero8();
#pragma unroll
    for (int nh = 0; nh < 4; ++nh) {
      s[nh] = wmma_f16(qf[0], kf[nh * 2 + 0], s[nh]);
      s[nh] = wmma_f16(qf[1], kf[nh * 2 + 1], s[nh]);
    }
    // ---- stash scaled scores (C-layout -> LDS rows)
#pragma unroll
    for (int nh = 0; nh < 4; ++nh)
#pragma unroll
      for (int r = 0; r < 8; ++r)
        Sf[wave][mb + r][nh * 16 + nl] = s[nh][r] * 0.125f;
    __syncthreads();
    // ---- online softmax: lane j (<16) owns row j of this wave's tile
    if (lane < 16) {
      float rmax = -3.0e38f;
#pragma unroll
      for (int j = 0; j < 64; ++j) rmax = fmaxf(rmax, Sf[wave][lane][j]);
      float mnew = fmaxf(m_run, rmax);
      float alpha = __expf(m_run - mnew);
      float sum = 0.0f;
#pragma unroll
      for (int j = 0; j < 64; ++j) {
        float p = __expf(Sf[wave][lane][j] - mnew);
        sum += p;
        Pf[wave][lane][j] = (_Float16)p;
      }
      l_run = l_run * alpha + sum;
      m_run = mnew;
      Alpha[wave][lane] = alpha;
    }
    __syncthreads();
    // ---- P as two A-fragments (16x64), batch-load V frags, rescale O, PV
    v16h pf[2];
#pragma unroll
    for (int c = 0; c < 2; ++c)
      pf[c] = load_afrag(&Pf[wave][nl][0] + 32 * c, ksel);
    v16h vf[8];
#pragma unroll
    for (int f = 0; f < 4; ++f)
#pragma unroll
      for (int c = 0; c < 2; ++c)
        vf[f * 2 + c] = *(const v16h*)(vtb + (size_t)(f * 16 + nl) * TSEQ +
                                       j0 + 32 * c + 16 * ksel);
    float ar[8];
#pragma unroll
    for (int r = 0; r < 8; ++r) ar[r] = Alpha[wave][mb + r];
#pragma unroll
    for (int f = 0; f < 4; ++f) {
#pragma unroll
      for (int r = 0; r < 8; ++r) o[f][r] *= ar[r];
      o[f] = wmma_f16(pf[0], vf[f * 2 + 0], o[f]);
      o[f] = wmma_f16(pf[1], vf[f * 2 + 1], o[f]);
    }
  }

  if (lane < 16) Linv[wave][lane] = 1.0f / l_run;
  __syncthreads();
  float lr[8];
#pragma unroll
  for (int r = 0; r < 8; ++r) lr[r] = Linv[wave][mb + r];
#pragma unroll
  for (int f = 0; f < 4; ++f)
#pragma unroll
    for (int r = 0; r < 8; ++r) {
      size_t row = (size_t)(b * TSEQ + qt * 16 + mb + r);
      ctx[row * DMODEL + h * HDIM + f * 16 + nl] = (_Float16)(o[f][r] * lr[r]);
    }
}

// ---------------------------------------------------------------------------
extern "C" void kernel_launch(void* const* d_in, const int* in_sizes, int n_in,
                              void* d_out, int out_size, void* d_ws, size_t ws_size,
                              hipStream_t stream) {
  const float* x  = (const float*)d_in[0];
  const float* Wq = (const float*)d_in[1];
  const float* Wk = (const float*)d_in[2];
  const float* Wv = (const float*)d_in[3];
  const float* Wo = (const float*)d_in[4];
  float* out = (float*)d_out;

  // workspace carve-up (f16 elements)
  _Float16* ws  = (_Float16*)d_ws;
  _Float16* xh  = ws;                               // 4096*2048
  _Float16* wqh = xh  + (size_t)ROWS * DMODEL;      // 2048*2048
  _Float16* wkh = wqh + (size_t)DMODEL * DMODEL;    // 512*2048
  _Float16* wvh = wkh + (size_t)KVDIM * DMODEL;     // 512*2048
  _Float16* woh = wvh + (size_t)KVDIM * DMODEL;     // 2048*2048
  _Float16* qh  = woh + (size_t)DMODEL * DMODEL;    // 4096*2048
  _Float16* kh  = qh  + (size_t)ROWS * DMODEL;      // 4096*512
  _Float16* vh  = kh  + (size_t)ROWS * KVDIM;       // 4096*512
  _Float16* vth = vh  + (size_t)ROWS * KVDIM;       // 4096*512 (transposed)
  _Float16* ctxh= vth + (size_t)ROWS * KVDIM;       // 4096*2048

  // 1) f32 -> f16 conversions
  cvt_f32_f16<<<4096, 256, 0, stream>>>(x,  xh,  (long long)ROWS * DMODEL);
  cvt_f32_f16<<<4096, 256, 0, stream>>>(Wq, wqh, (long long)DMODEL * DMODEL);
  cvt_f32_f16<<<2048, 256, 0, stream>>>(Wk, wkh, (long long)KVDIM * DMODEL);
  cvt_f32_f16<<<2048, 256, 0, stream>>>(Wv, wvh, (long long)KVDIM * DMODEL);
  cvt_f32_f16<<<4096, 256, 0, stream>>>(Wo, woh, (long long)DMODEL * DMODEL);

  // 2) projections: Q = x Wq^T, K = x Wk^T, V = x Wv^T
  gemm_f16_wmma<_Float16><<<dim3(DMODEL / 64, ROWS / 128), 128, 0, stream>>>(
      xh, wqh, qh, ROWS, DMODEL, DMODEL);
  gemm_f16_wmma<_Float16><<<dim3(KVDIM / 64, ROWS / 128), 128, 0, stream>>>(
      xh, wkh, kh, ROWS, KVDIM, DMODEL);
  gemm_f16_wmma<_Float16><<<dim3(KVDIM / 64, ROWS / 128), 128, 0, stream>>>(
      xh, wvh, vh, ROWS, KVDIM, DMODEL);

  // 3) V transpose per KV head
  {
    long long n = (long long)ROWS * KVDIM;
    vtrans_kernel<<<(unsigned)((n + 255) / 256), 256, 0, stream>>>(vh, vth);
  }

  // 4) flash attention -> ctx (f16)
  gqa_attn<<<2048, 128, 0, stream>>>(qh, kh, vth, ctxh);

  // 5) out = ctx Wo^T (f32 output)
  gemm_f16_wmma<float><<<dim3(DMODEL / 64, ROWS / 128), 128, 0, stream>>>(
      ctxh, woh, out, ROWS, DMODEL, DMODEL);
}
